// NestedEventMamba_21921513079117
// MI455X (gfx1250) — matrix operand order repaired and checked by reading, not compiled, not run on hardware
//
#include <hip/hip_runtime.h>
#include <math.h>

typedef __attribute__((ext_vector_type(16))) _Float16 v16h;
typedef __attribute__((ext_vector_type(8)))  _Float16 v8h;
typedef __attribute__((ext_vector_type(8)))  float    v8f;

// ---------------- device math helpers ----------------
__device__ __forceinline__ float gelu_f(float x) {
  const float c = 0.7978845608028654f; // sqrt(2/pi)
  return 0.5f * x * (1.0f + tanhf(c * (x + 0.044715f * x * x * x)));
}
__device__ __forceinline__ float silu_f(float x) {
  return x / (1.0f + __expf(-x));
}
__device__ __forceinline__ float softplus_f(float x) {
  return (x > 20.0f) ? x : log1pf(__expf(x));
}

// ---------------- operand packing (fp32 -> f16, zero padded) ----------------
// A: [M, K] (row stride lda) -> [M, Kpad] f16
__global__ __launch_bounds__(256) void pack_a16_kernel(
    const float* __restrict__ A, int lda, _Float16* __restrict__ O,
    long long total, int K, int Kpad)
{
  long long idx = (long long)blockIdx.x * blockDim.x + threadIdx.x;
  if (idx >= total) return;
  int k = (int)(idx % Kpad);
  long long m = idx / Kpad;
  O[idx] = (_Float16)((k < K) ? A[m * lda + k] : 0.0f);
}

// W: [N, K] (row stride ldw) -> [Npad, Kpad] f16 (rows >= N zeroed)
__global__ __launch_bounds__(256) void pack_w16_kernel(
    const float* __restrict__ W, int ldw, _Float16* __restrict__ O,
    long long total, int N, int K, int Kpad)
{
  long long idx = (long long)blockIdx.x * blockDim.x + threadIdx.x;
  if (idx >= total) return;
  int k = (int)(idx % Kpad);
  long long n = idx / Kpad;
  O[idx] = (_Float16)((n < (long long)N && k < K) ? W[n * ldw + k] : 0.0f);
}

// ---------------- WMMA GEMM: one wave -> 16x64 tile ----------------
// C[M,N] = act(A[M,K] @ W[N,K]^T + bias[N]) (+ Res[M,N])
// A16: [M, Kpad] f16 (Kpad % 32 == 0), W16: [Npad, Kpad] f16 (Npad % 64 == 0)
// Branch-free operand loads: two b128 loads per fragment, 4 WMMAs per K-step
// on independent accumulators. EXEC stays all-1s inside active waves.
// Fragment layouts used here (contiguous k, same order for A and B):
//   A (16x32 f16): lane l    -> row=l&15, K=(l>>4)*16 + {0..15}
//   B (32x16 f16): lane l    -> col=l&15, K=(l>>4)*16 + {0..15}
//   C/D (16x16 f32): vgpr r -> row = r + (l>>4)*8, col = l&15
// act: 0=none 1=gelu 2=silu 3=softplus
__global__ __launch_bounds__(256) void gemm_wmma_kernel(
    const _Float16* __restrict__ A16, int Kpad,
    const _Float16* __restrict__ W16,
    const float* __restrict__ bias,
    float* __restrict__ C, int ldc,
    const float* __restrict__ Res, int ldr,
    int M, int N, int act, int tm, int tg)
{
  int wid = blockIdx.x * (blockDim.x >> 5) + (threadIdx.x >> 5);
  if (wid >= tm * tg) return;           // uniform whole-wave exit
  int g   = wid % tg;
  int tmm = wid / tg;
  int m0 = tmm << 4;
  int n0 = g << 6;                      // 64 columns per wave
  int lane = threadIdx.x & 31;
  int half = lane >> 4;
  int l15  = lane & 15;

  const _Float16* arow = A16 + (long long)(m0 + l15) * Kpad;
  v8f acc0 = {}, acc1 = {}, acc2 = {}, acc3 = {};

  for (int k0 = 0; k0 < Kpad; k0 += 32) {
    // A fragment: K = {k0+half*16 .. +15}
    v8h alo = *(const v8h*)(arow + k0 + half * 16);
    v8h ahi = *(const v8h*)(arow + k0 + half * 16 + 8);
    v16h av;
    #pragma unroll
    for (int j = 0; j < 8; ++j) { av[j] = alo[j]; av[j + 8] = ahi[j]; }

    #pragma unroll
    for (int t = 0; t < 4; ++t) {
      const _Float16* wrow = W16 + (long long)(n0 + t * 16 + l15) * Kpad + k0 + half * 16;
      v8h b0 = *(const v8h*)(wrow);
      v8h b1 = *(const v8h*)(wrow + 8);
      v16h bv;
      #pragma unroll
      for (int j = 0; j < 8; ++j) { bv[j] = b0[j]; bv[j + 8] = b1[j]; }
      if (t == 0) acc0 = __builtin_amdgcn_wmma_f32_16x16x32_f16(false, av, false, bv, (short)0, acc0, false, false);
      if (t == 1) acc1 = __builtin_amdgcn_wmma_f32_16x16x32_f16(false, av, false, bv, (short)0, acc1, false, false);
      if (t == 2) acc2 = __builtin_amdgcn_wmma_f32_16x16x32_f16(false, av, false, bv, (short)0, acc2, false, false);
      if (t == 3) acc3 = __builtin_amdgcn_wmma_f32_16x16x32_f16(false, av, false, bv, (short)0, acc3, false, false);
    }
  }

  #pragma unroll
  for (int t = 0; t < 4; ++t) {
    v8f acc = (t == 0) ? acc0 : (t == 1) ? acc1 : (t == 2) ? acc2 : acc3;
    int n = n0 + t * 16 + l15;
    if (n >= N) continue;
    float bv = bias ? bias[n] : 0.0f;
    #pragma unroll
    for (int r = 0; r < 8; ++r) {
      int m = m0 + r + half * 8;
      if (m < M) {
        float v = acc[r] + bv;
        if (act == 1)      v = gelu_f(v);
        else if (act == 2) v = silu_f(v);
        else if (act == 3) v = softplus_f(v);
        if (Res) v += Res[(long long)m * ldr + n];
        C[(long long)m * ldc + n] = v;
      }
    }
  }
}

// ---------------- LayerNorm: one wave per token (D=128) ----------------
__global__ __launch_bounds__(256) void ln_kernel(
    const float* __restrict__ X, const float* __restrict__ g,
    const float* __restrict__ b, float* __restrict__ O, int M)
{
  int wv = blockIdx.x * (blockDim.x >> 5) + (threadIdx.x >> 5);
  if (wv >= M) return;
  int lane = threadIdx.x & 31;
  const float* row = X + (long long)wv * 128;
  float v0 = row[lane], v1 = row[lane + 32], v2 = row[lane + 64], v3 = row[lane + 96];
  float s  = v0 + v1 + v2 + v3;
  float s2 = v0 * v0 + v1 * v1 + v2 * v2 + v3 * v3;
  for (int o = 16; o > 0; o >>= 1) {
    s  += __shfl_xor(s,  o, 32);
    s2 += __shfl_xor(s2, o, 32);
  }
  float m   = s * (1.0f / 128.0f);
  float inv = rsqrtf(s2 * (1.0f / 128.0f) - m * m + 1e-5f);
  float* orow = O + (long long)wv * 128;
  orow[lane]      = (v0 - m) * inv * g[lane]      + b[lane];
  orow[lane + 32] = (v1 - m) * inv * g[lane + 32] + b[lane + 32];
  orow[lane + 64] = (v2 - m) * inv * g[lane + 64] + b[lane + 64];
  orow[lane + 96] = (v3 - m) * inv * g[lane + 96] + b[lane + 96];
}

// ---------------- u = dwconv3(h) + h (per-sequence, zero pad) ----------------
__global__ __launch_bounds__(256) void dwconv_kernel(
    const float* __restrict__ H, const float* __restrict__ w,
    const float* __restrict__ cb, float* __restrict__ U, int M, int L)
{
  long long idx = (long long)blockIdx.x * blockDim.x + threadIdx.x;
  if (idx >= (long long)M * 128) return;
  int d = (int)(idx & 127);
  long long token = idx >> 7;
  int tl = (int)(token % L);
  float hc = H[token * 128 + d];
  float hm = (tl > 0)      ? H[(token - 1) * 128 + d] : 0.0f;
  float hp = (tl < L - 1)  ? H[(token + 1) * 128 + d] : 0.0f;
  U[token * 128 + d] = hm * w[d] + hc * w[128 + d] + hp * w[256 + d] + cb[d] + hc;
}

// ---------------- xc = silu(causal conv4(x-part of xz)) ----------------
__global__ __launch_bounds__(256) void conv4_silu_kernel(
    const float* __restrict__ XZ, const float* __restrict__ w,
    const float* __restrict__ cb, float* __restrict__ XC, int M, int L)
{
  long long idx = (long long)blockIdx.x * blockDim.x + threadIdx.x;
  if (idx >= (long long)M * 256) return;
  int d = (int)(idx & 255);
  long long token = idx >> 8;
  int tl = (int)(token % L);
  long long s0 = token - tl;  // sequence start token
  float a = cb[d];
  #pragma unroll
  for (int j = 0; j < 4; ++j) {
    int tt = tl - 3 + j;
    if (tt >= 0) a += w[j * 256 + d] * XZ[(s0 + tt) * 512 + d];
  }
  XC[token * 256 + d] = silu_f(a);
}

// ---------------- selective scan: one block = one sequence ----------------
__global__ __launch_bounds__(256) void scan_kernel(
    const float* __restrict__ DT, const float* __restrict__ XC,
    const float* __restrict__ DBL, const float* __restrict__ A_log,
    float* __restrict__ YS, int L)
{
  __shared__ float sB[256 * 16];
  __shared__ float sC[256 * 16];
  int seq = blockIdx.x;
  long long tok0 = (long long)seq * L;
  int tid = threadIdx.x;  // channel d (din = 256)
  for (int i = tid; i < L * 32; i += 256) {
    int t = i >> 5, j = i & 31;
    float v = DBL[(tok0 + t) * 40 + 8 + j];
    if (j < 16) sB[t * 16 + j] = v;
    else        sC[t * 16 + (j - 16)] = v;
  }
  __syncthreads();
  float Ad[16];
  #pragma unroll
  for (int n = 0; n < 16; ++n) Ad[n] = -__expf(A_log[tid * 16 + n]);
  float h[16];
  #pragma unroll
  for (int n = 0; n < 16; ++n) h[n] = 0.0f;
  for (int t = 0; t < L; ++t) {
    float dt = DT[(tok0 + t) * 256 + tid];
    float xv = XC[(tok0 + t) * 256 + tid];
    float dx = dt * xv;
    float y = 0.0f;
    #pragma unroll
    for (int n = 0; n < 16; ++n) {
      float e = __expf(dt * Ad[n]);
      h[n] = e * h[n] + dx * sB[t * 16 + n];
      y += h[n] * sC[t * 16 + n];
    }
    YS[(tok0 + t) * 256 + tid] = y;
  }
}

// ---------------- y = (ys + xc*D) * silu(z) ----------------
__global__ __launch_bounds__(256) void gate_kernel(
    const float* __restrict__ YS, const float* __restrict__ XC,
    const float* __restrict__ XZ, const float* __restrict__ Dp,
    float* __restrict__ Y, int M)
{
  long long idx = (long long)blockIdx.x * blockDim.x + threadIdx.x;
  if (idx >= (long long)M * 256) return;
  int d = (int)(idx & 255);
  long long token = idx >> 8;
  float z = XZ[token * 512 + 256 + d];
  Y[idx] = (YS[idx] + XC[idx] * Dp[d]) * silu_f(z);
}

// ---------------- segment pooling: win = 0.5*(mean + max) over t<slen ----------------
__global__ __launch_bounds__(128) void pool_kernel(
    const float* __restrict__ X, const int* __restrict__ seglen,
    float* __restrict__ OUT, int L)
{
  int seg = blockIdx.x, d = threadIdx.x;
  int sl = seglen[seg]; if (sl < 1) sl = 1; if (sl > L) sl = L;
  const float* base = X + (long long)seg * L * 128 + d;
  float s = 0.0f, mx = -3.402823466e38f;
  for (int t = 0; t < sl; ++t) {
    float v = base[(long long)t * 128];
    s += v; mx = fmaxf(mx, v);
  }
  OUT[seg * 128 + d] = 0.5f * (s / (float)sl + mx);
}

// ---------------- sequence pooling: g = mean over s<qlen ----------------
__global__ __launch_bounds__(128) void seqpool_kernel(
    const float* __restrict__ SEQ, const int* __restrict__ qlenv,
    float* __restrict__ G)
{
  int b = blockIdx.x, d = threadIdx.x;
  int ql = qlenv[b]; if (ql < 1) ql = 1; if (ql > 32) ql = 32;
  float s = 0.0f;
  for (int t = 0; t < ql; ++t) s += SEQ[(b * 32 + t) * 128 + d];
  G[b * 128 + d] = s / (float)ql;
}

// ---------------- classifier head: LN -> gelu(128->64) -> 64->11 ----------------
__global__ __launch_bounds__(128) void head_kernel(
    const float* __restrict__ G, const float* __restrict__ lg,
    const float* __restrict__ lb, const float* __restrict__ w1,
    const float* __restrict__ b1, const float* __restrict__ w2,
    const float* __restrict__ b2, float* __restrict__ OUT)
{
  __shared__ float sh[128];
  __shared__ float hid[64];
  __shared__ float red[2];
  int b = blockIdx.x, d = threadIdx.x;
  float v = G[b * 128 + d];
  sh[d] = v;
  __syncthreads();
  if (d == 0) {
    float s = 0.0f, s2 = 0.0f;
    for (int i = 0; i < 128; ++i) { s += sh[i]; s2 += sh[i] * sh[i]; }
    float m = s / 128.0f;
    red[0] = m;
    red[1] = rsqrtf(s2 / 128.0f - m * m + 1e-5f);
  }
  __syncthreads();
  float nv = (v - red[0]) * red[1] * lg[d] + lb[d];
  sh[d] = nv;
  __syncthreads();
  if (d < 64) {
    float a = b1[d];
    for (int i = 0; i < 128; ++i) a += w1[d * 128 + i] * sh[i];
    hid[d] = gelu_f(a);
  }
  __syncthreads();
  if (d < 11) {
    float a = b2[d];
    for (int i = 0; i < 64; ++i) a += w2[d * 64 + i] * hid[i];
    OUT[b * 11 + d] = a;
  }
}

// ---------------- host side ----------------
static inline int cdiv_i(int a, int b) { return (a + b - 1) / b; }
static inline long long cdiv_ll(long long a, long long b) { return (a + b - 1) / b; }

struct BlockP {
  const float *ln1_g, *ln1_b, *lconv_w, *lconv_b, *ln2_g, *ln2_b;
  const float *mlp_w1, *mlp_b1, *mlp_w2, *mlp_b2;
  const float *in_proj, *conv_w, *conv_b, *x_proj, *dt_w, *dt_b, *A_log, *Dp, *out_proj;
};

extern "C" void kernel_launch(void* const* d_in, const int* in_sizes, int n_in,
                              void* d_out, int out_size, void* d_ws, size_t ws_size,
                              hipStream_t stream) {
  (void)in_sizes; (void)n_in; (void)out_size; (void)ws_size;

  // ---- inputs in setup_inputs() insertion order ----
  const float* feats   = (const float*)d_in[0];  // (8,32,256,130)
  const int*   seg_len = (const int*)  d_in[1];  // (8,32)
  const int*   seq_len = (const int*)  d_in[2];  // (8,)
  int p = 3;
  auto F = [&](int i) { return (const float*)d_in[i]; };
  const float* ad_w1 = F(p++); const float* ad_b1 = F(p++);
  const float* ad_w2 = F(p++); const float* ad_b2 = F(p++);
  BlockP blk[4];  // intra0, intra1, inter0, inter1
  for (int bi = 0; bi < 4; ++bi) {
    BlockP& q = blk[bi];
    q.ln1_g  = F(p++); q.ln1_b  = F(p++); q.lconv_w = F(p++); q.lconv_b = F(p++);
    q.ln2_g  = F(p++); q.ln2_b  = F(p++);
    q.mlp_w1 = F(p++); q.mlp_b1 = F(p++); q.mlp_w2  = F(p++); q.mlp_b2  = F(p++);
    q.in_proj = F(p++); q.conv_w = F(p++); q.conv_b = F(p++); q.x_proj = F(p++);
    q.dt_w = F(p++); q.dt_b = F(p++); q.A_log = F(p++); q.Dp = F(p++); q.out_proj = F(p++);
  }
  const float* head_ln_g = F(p++); const float* head_ln_b = F(p++);
  const float* head_w1 = F(p++); const float* head_b1 = F(p++);
  const float* head_w2 = F(p++); const float* head_b2 = F(p++);

  // ---- workspace carve-up (all buffers fully written before read) ----
  const int M1 = 8 * 32 * 256;  // 65536 intra tokens
  const int M2 = 8 * 32;        // 256 inter tokens
  char* wsb = (char*)d_ws;
  size_t off = 0;
  auto allocB = [&](size_t bytes) -> void* {
    void* q = (void*)(wsb + off);
    off += (bytes + 255) & ~(size_t)255;
    return q;
  };
  float* x      = (float*)allocB((size_t)M1 * 128 * 4);  // residual stream
  float* bufH   = (float*)allocB((size_t)M1 * 128 * 4);  // LN outputs
  float* bufU   = (float*)allocB((size_t)M1 * 128 * 4);  // dwconv output
  float* buf512 = (float*)allocB((size_t)M1 * 512 * 4);  // xz / mlp hidden
  float* bufXC  = (float*)allocB((size_t)M1 * 256 * 4);  // conv4+silu output
  float* bufDT  = (float*)allocB((size_t)M1 * 256 * 4);  // dt, then gated y
  float* bufYS  = (float*)allocB((size_t)M1 * 256 * 4);  // scan output
  float* bufDBL = (float*)allocB((size_t)M1 * 40 * 4);   // x_proj output
  float* seqb   = (float*)allocB((size_t)M2 * 128 * 4);  // inter residual
  float* gbuf   = (float*)allocB((size_t)8 * 128 * 4);   // pooled per-batch
  _Float16* a16 = (_Float16*)allocB((size_t)M1 * 512 * 2);  // packed A
  _Float16* w16 = (_Float16*)allocB((size_t)512 * 512 * 2); // packed W

  // GEMM with f16 packing. NT=4 (16x64 tile per wave), Npad -> mult of 64.
  auto gemm = [&](const float* A, int lda, const float* W, int ldw,
                  const float* bias, float* C, int ldc,
                  const float* Res, int ldr, int M, int N, int K, int act) {
    int Kpad = cdiv_i(K, 32) * 32;
    int Npad = cdiv_i(N, 64) * 64;
    long long na = (long long)M * Kpad;
    pack_a16_kernel<<<(int)cdiv_ll(na, 256), 256, 0, stream>>>(A, lda, a16, na, K, Kpad);
    long long nw = (long long)Npad * Kpad;
    pack_w16_kernel<<<(int)cdiv_ll(nw, 256), 256, 0, stream>>>(W, ldw, w16, nw, N, K, Kpad);
    int tm = M / 16;             // M is always a multiple of 16 here
    int tg = Npad / 64;
    int blocks = cdiv_i(tm * tg, 8);
    gemm_wmma_kernel<<<blocks, 256, 0, stream>>>(a16, Kpad, w16, bias, C, ldc,
                                                 Res, ldr, M, N, act, tm, tg);
  };

  auto run_block = [&](float* X, int M, int L, const BlockP& q) {
    ln_kernel<<<cdiv_i(M, 8), 256, 0, stream>>>(X, q.ln1_g, q.ln1_b, bufH, M);
    dwconv_kernel<<<cdiv_i(M * 128, 256), 256, 0, stream>>>(bufH, q.lconv_w, q.lconv_b, bufU, M, L);
    gemm(bufU, 128, q.in_proj, 128, nullptr, buf512, 512, nullptr, 0, M, 512, 128, 0);
    conv4_silu_kernel<<<cdiv_i(M * 256, 256), 256, 0, stream>>>(buf512, q.conv_w, q.conv_b, bufXC, M, L);
    gemm(bufXC, 256, q.x_proj, 256, nullptr, bufDBL, 40, nullptr, 0, M, 40, 256, 0);
    gemm(bufDBL, 40, q.dt_w, 8, q.dt_b, bufDT, 256, nullptr, 0, M, 256, 8, 3);
    scan_kernel<<<M / L, 256, 0, stream>>>(bufDT, bufXC, bufDBL, q.A_log, bufYS, L);
    gate_kernel<<<cdiv_i(M * 256, 256), 256, 0, stream>>>(bufYS, bufXC, buf512, q.Dp, bufDT, M);
    gemm(bufDT, 256, q.out_proj, 256, nullptr, X, 128, X, 128, M, 128, 256, 0);
    ln_kernel<<<cdiv_i(M, 8), 256, 0, stream>>>(X, q.ln2_g, q.ln2_b, bufH, M);
    gemm(bufH, 128, q.mlp_w1, 128, q.mlp_b1, buf512, 512, nullptr, 0, M, 512, 128, 1);
    gemm(buf512, 512, q.mlp_w2, 512, q.mlp_b2, X, 128, X, 128, M, 128, 512, 0);
  };

  // ---- adapter ----
  gemm(feats, 130, ad_w1, 130, ad_b1, bufH, 128, nullptr, 0, M1, 128, 130, 1);
  gemm(bufH, 128, ad_w2, 128, ad_b2, x, 128, nullptr, 0, M1, 128, 128, 0);

  // ---- intra blocks (256 sequences of length 256) ----
  run_block(x, M1, 256, blk[0]);
  run_block(x, M1, 256, blk[1]);

  // ---- segment pooling -> (8,32,128) ----
  pool_kernel<<<256, 128, 0, stream>>>(x, seg_len, seqb, 256);

  // ---- inter blocks (8 sequences of length 32) ----
  run_block(seqb, M2, 32, blk[2]);
  run_block(seqb, M2, 32, blk[3]);

  // ---- sequence pooling + head ----
  seqpool_kernel<<<8, 128, 0, stream>>>(seqb, seq_len, gbuf);
  head_kernel<<<8, 128, 0, stream>>>(gbuf, head_ln_g, head_ln_b,
                                     head_w1, head_b1, head_w2, head_b2,
                                     (float*)d_out);
}
